// SelfAttention_64750926955049
// MI455X (gfx1250) — matrix operand order repaired and checked by reading, not compiled, not run on hardware
//
#include <hip/hip_runtime.h>
#include <hip/hip_bf16.h>

typedef __attribute__((ext_vector_type(16))) _Float16 v16h;
typedef __attribute__((ext_vector_type(8)))  _Float16 v8h;
typedef __attribute__((ext_vector_type(8)))  float    v8f;

#define NB   2
#define NS   2048
#define NDM  1024
#define NH   16
#define ND   64

static __device__ __forceinline__ v8f wmma16(v16h a, v16h b, v8f c) {
  // D = A(16x32 f16) * B(32x16 f16) + C(16x16 f32)
  return __builtin_amdgcn_wmma_f32_16x16x32_f16(false, a, false, b, (short)0, c,
                                                false, false);
}

// Convert 16 consecutive floats (global or LDS) to a WMMA f16 operand.
static __device__ __forceinline__ v16h cvt16(const float* __restrict__ p) {
  const float4 f0 = ((const float4*)p)[0];
  const float4 f1 = ((const float4*)p)[1];
  const float4 f2 = ((const float4*)p)[2];
  const float4 f3 = ((const float4*)p)[3];
  v16h a;
  a[0]  = (_Float16)f0.x; a[1]  = (_Float16)f0.y;
  a[2]  = (_Float16)f0.z; a[3]  = (_Float16)f0.w;
  a[4]  = (_Float16)f1.x; a[5]  = (_Float16)f1.y;
  a[6]  = (_Float16)f1.z; a[7]  = (_Float16)f1.w;
  a[8]  = (_Float16)f2.x; a[9]  = (_Float16)f2.y;
  a[10] = (_Float16)f2.z; a[11] = (_Float16)f2.w;
  a[12] = (_Float16)f3.x; a[13] = (_Float16)f3.y;
  a[14] = (_Float16)f3.z; a[15] = (_Float16)f3.w;
  return a;
}

// Build a v16h A-operand from two 8-float chunks (per-lane K-chunks).
static __device__ __forceinline__ v16h cvt8x2(const float* __restrict__ lo,
                                              const float* __restrict__ hi) {
  const float4 f0 = ((const float4*)lo)[0];
  const float4 f1 = ((const float4*)lo)[1];
  const float4 f2 = ((const float4*)hi)[0];
  const float4 f3 = ((const float4*)hi)[1];
  v16h a;
  a[0]  = (_Float16)f0.x; a[1]  = (_Float16)f0.y;
  a[2]  = (_Float16)f0.z; a[3]  = (_Float16)f0.w;
  a[4]  = (_Float16)f1.x; a[5]  = (_Float16)f1.y;
  a[6]  = (_Float16)f1.z; a[7]  = (_Float16)f1.w;
  a[8]  = (_Float16)f2.x; a[9]  = (_Float16)f2.y;
  a[10] = (_Float16)f2.z; a[11] = (_Float16)f2.w;
  a[12] = (_Float16)f3.x; a[13] = (_Float16)f3.y;
  a[14] = (_Float16)f3.z; a[15] = (_Float16)f3.w;
  return a;
}

// ---------------------------------------------------------------------------
// Kernel 1: projection GEMM.  out = X[4096,1024] @ W[1024,1024] -> f16
// grid = (64, 64, 3): x = 64-row block, y = 16-col tile, z selects Q/K/V.
// block = 128 threads = 4 waves; each wave owns a 16x16 output tile, the 4
// waves stack in M and share a K=64 W-tile in LDS. The W tile is staged
// TRANSPOSED via per-lane GLOBAL_LOAD_ASYNC_TO_LDS_B32 (async scatter does
// the transpose for free, tracked on ASYNCcnt).
// Q,K stored [B,H,S,D]; V stored transposed [B,H,D,S] for the PV matmul.
// ---------------------------------------------------------------------------
__global__ __launch_bounds__(128) void proj_qkv(
    const float* __restrict__ Qin, const float* __restrict__ Kin,
    const float* __restrict__ Vin,
    const float* __restrict__ WQ, const float* __restrict__ WK,
    const float* __restrict__ WV,
    _Float16* __restrict__ qh, _Float16* __restrict__ kh,
    _Float16* __restrict__ vT)
{
  const int z = blockIdx.z;
  const float* __restrict__ X = (z == 0) ? Qin : (z == 1) ? Kin : Vin;
  const float* __restrict__ W = (z == 0) ? WQ  : (z == 1) ? WK  : WV;

  __shared__ __attribute__((aligned(16))) float sW[16 * 64]; // [n][k], fp32

  const int t    = threadIdx.x;
  const int lane = t & 31;
  const int wave = t >> 5;
  const int m0   = blockIdx.x * 64;
  const int n0   = blockIdx.y * 16;
  const int m    = lane & 15;      // row within tile (A) / col within tile (B,C)
  const int h2   = lane >> 4;      // half-wave select
  const int n    = m;              // B/C column index
  const int row  = m0 + wave * 16 + m;

  const int wn = t & 15;           // cooperative W-load: column
  const int wk = t >> 4;           // cooperative W-load: k-row 0..7

  v8f c = {0.f, 0.f, 0.f, 0.f, 0.f, 0.f, 0.f, 0.f};

  for (int k0 = 0; k0 < NDM; k0 += 64) {
    __syncthreads();  // previous iteration's LDS reads are done
    // Async-stage W[k0..k0+63, n0..n0+15] into LDS transposed as sW[n][k].
    // Per-lane LDS destination address => the DMA performs the transpose.
#pragma unroll
    for (int i = 0; i < 8; ++i) {
      const int kk = wk + 8 * i;
      const unsigned ldsa =
          (unsigned)(size_t)(&sW[wn * 64 + kk]);
      const float* __restrict__ g = &W[(size_t)(k0 + kk) * NDM + n0 + wn];
      asm volatile("global_load_async_to_lds_b32 %0, %1, off"
                   :: "v"(ldsa), "v"(g) : "memory");
    }
    asm volatile("s_wait_asynccnt 0x0" ::: "memory");
    __syncthreads();

    // A operands: per-lane K-chunks [16h, 16h+8) and [16h+8, 16h+16).
    const float* __restrict__ xr = X + (size_t)row * NDM + k0;
    const v16h a0 = cvt8x2(xr + 16 * h2, xr + 8 + 16 * h2);
    const v16h a1 = cvt8x2(xr + 32 + 16 * h2, xr + 40 + 16 * h2);

    // B operands: lane holds column n, K = [16h, 16h+16) -> 64B contiguous.
    const v16h b0 = cvt16(&sW[n * 64 + 16 * h2]);
    const v16h b1 = cvt16(&sW[n * 64 + 32 + 16 * h2]);

    c = wmma16(a0, b0, c);
    c = wmma16(a1, b1, c);
  }

  // C layout: VGPR r <-> output row (r + 8*h2), column n.
  const int col   = n0 + n;
  const int hh    = col >> 6;        // head
  const int d     = col & (ND - 1);  // dim within head
  const int rbase = m0 + wave * 16 + 8 * h2;
  const int bb    = rbase >> 11;     // batch (rows are b*NS + s)
  const int sbase = rbase & (NS - 1);

  if (z != 2) {
    _Float16* __restrict__ O = (z == 0) ? qh : kh;
#pragma unroll
    for (int r = 0; r < 8; ++r) {
      O[((((size_t)bb * NH + hh) * NS) + sbase + r) * ND + d] = (_Float16)c[r];
    }
  } else {
    // Transposed store [B,H,D,S]: 8 consecutive s -> one contiguous b128.
    v8h o;
#pragma unroll
    for (int r = 0; r < 8; ++r) o[r] = (_Float16)c[r];
    *(v8h*)(&vT[(((size_t)bb * NH + hh) * ND + d) * NS + sbase]) = o;
  }
}

// ---------------------------------------------------------------------------
// Kernel 2: flash attention forward, one wave per (b, h, 16-query tile).
// block = 128 threads = 4 waves; grid = 4096/4 = 1024 blocks.
// Per 32-key block: 4 score WMMAs + online softmax + 4 output WMMAs.
// ---------------------------------------------------------------------------
__global__ __launch_bounds__(128) void attn_fwd(
    const _Float16* __restrict__ qh, const _Float16* __restrict__ kh,
    const _Float16* __restrict__ vT, const int* __restrict__ vlen_p,
    float* __restrict__ out)
{
  __shared__ __attribute__((aligned(32))) _Float16 sP[4][16 * 32];

  const int t    = threadIdx.x;
  const int lane = t & 31;
  const int wave = t >> 5;
  const int tile = blockIdx.x * 4 + wave;
  const int b    = tile >> 11;          // 2048 tiles per batch
  const int hh   = (tile >> 7) & 15;    // 128 tiles per head
  const int q0   = (tile & 127) * 16;
  const int m    = lane & 15;
  const int h2   = lane >> 4;
  const int n    = m;
  const int vlen = vlen_p[b];

  // Q tile -> A operands for the two K=32 d-chunks.
  const _Float16* __restrict__ qr =
      qh + (((size_t)b * NH + hh) * NS + q0 + m) * ND;
  const v8h q00 = *(const v8h*)(qr + 16 * h2);
  const v8h q01 = *(const v8h*)(qr + 8 + 16 * h2);
  const v8h q10 = *(const v8h*)(qr + 32 + 16 * h2);
  const v8h q11 = *(const v8h*)(qr + 40 + 16 * h2);
  const v16h a0 = __builtin_shufflevector(q00, q01, 0, 1, 2, 3, 4, 5, 6, 7,
                                          8, 9, 10, 11, 12, 13, 14, 15);
  const v16h a1 = __builtin_shufflevector(q10, q11, 0, 1, 2, 3, 4, 5, 6, 7,
                                          8, 9, 10, 11, 12, 13, 14, 15);

  float mrow[8], lrow[8];
  v8f acc[4];
#pragma unroll
  for (int r = 0; r < 8; ++r) { mrow[r] = -INFINITY; lrow[r] = 0.f; }
#pragma unroll
  for (int j = 0; j < 4; ++j)
#pragma unroll
    for (int r = 0; r < 8; ++r) acc[j][r] = 0.f;

  const _Float16* __restrict__ kbase = kh + (((size_t)b * NH + hh) * NS) * ND;
  const _Float16* __restrict__ vbase = vT + (((size_t)b * NH + hh) * ND) * NS;
  _Float16* __restrict__ myP = sP[wave];

  for (int kb = 0; kb < NS; kb += 32) {
    if (kb + 32 < NS)  // gfx1250 global_prefetch of next K block
      __builtin_prefetch(kbase + (size_t)(kb + 32 + n) * ND, 0, 1);

    // K tiles as B operands: lane = (key col n, d chunk 16h..16h+15).
    const _Float16* __restrict__ kr0 = kbase + (size_t)(kb + n) * ND;
    const _Float16* __restrict__ kr1 = kr0 + 16 * ND;
    const v16h bk0a = *(const v16h*)(kr0 + 16 * h2);
    const v16h bk0b = *(const v16h*)(kr0 + 32 + 16 * h2);
    const v16h bk1a = *(const v16h*)(kr1 + 16 * h2);
    const v16h bk1b = *(const v16h*)(kr1 + 32 + 16 * h2);

    v8f s0 = {0.f, 0.f, 0.f, 0.f, 0.f, 0.f, 0.f, 0.f};
    v8f s1 = {0.f, 0.f, 0.f, 0.f, 0.f, 0.f, 0.f, 0.f};
    s0 = wmma16(a0, bk0a, s0);
    s0 = wmma16(a1, bk0b, s0);
    s1 = wmma16(a0, bk1a, s1);
    s1 = wmma16(a1, bk1b, s1);

    // Scale + key-padding mask (mask is per-column == per-lane uniform).
    const bool valid0 = (kb + n) < vlen;
    const bool valid1 = (kb + 16 + n) < vlen;
    float f0[8], f1[8], rmax[8];
#pragma unroll
    for (int r = 0; r < 8; ++r) {
      f0[r] = valid0 ? s0[r] * 0.125f : -1e9f;
      f1[r] = valid1 ? s1[r] * 0.125f : -1e9f;
      rmax[r] = fmaxf(f0[r], f1[r]);
    }
    // Row max across the 16 lanes of each half-wave.
#pragma unroll
    for (int off = 1; off < 16; off <<= 1)
#pragma unroll
      for (int r = 0; r < 8; ++r)
        rmax[r] = fmaxf(rmax[r], __shfl_xor(rmax[r], off, 32));

    float p0[8], p1[8], rs[8], sc[8];
#pragma unroll
    for (int r = 0; r < 8; ++r) {
      const float mn = fmaxf(mrow[r], rmax[r]);
      sc[r]  = __expf(mrow[r] - mn);
      mrow[r] = mn;
      p0[r]  = __expf(f0[r] - mn);
      p1[r]  = __expf(f1[r] - mn);
      rs[r]  = p0[r] + p1[r];
    }
#pragma unroll
    for (int off = 1; off < 16; off <<= 1)
#pragma unroll
      for (int r = 0; r < 8; ++r) rs[r] += __shfl_xor(rs[r], off, 32);
#pragma unroll
    for (int r = 0; r < 8; ++r) lrow[r] = lrow[r] * sc[r] + rs[r];
#pragma unroll
    for (int j = 0; j < 4; ++j)
#pragma unroll
      for (int r = 0; r < 8; ++r) acc[j][r] *= sc[r];

    // Transpose P (C layout -> A layout) through per-wave LDS buffer.
    __syncthreads();
#pragma unroll
    for (int r = 0; r < 8; ++r) {
      myP[(r + 8 * h2) * 32 + n]      = (_Float16)p0[r];
      myP[(r + 8 * h2) * 32 + 16 + n] = (_Float16)p1[r];
    }
    __syncthreads();
    const v8h pl = *(const v8h*)(myP + m * 32 + 16 * h2);
    const v8h ph = *(const v8h*)(myP + m * 32 + 8 + 16 * h2);
    const v16h pa = __builtin_shufflevector(pl, ph, 0, 1, 2, 3, 4, 5, 6, 7,
                                            8, 9, 10, 11, 12, 13, 14, 15);

    // Output WMMAs: B = transposed V, contiguous 32B per lane.
#pragma unroll
    for (int j = 0; j < 4; ++j) {
      const _Float16* __restrict__ vr =
          vbase + (size_t)(16 * j + n) * NS + kb + 16 * h2;
      const v16h bv = *(const v16h*)vr;
      acc[j] = wmma16(pa, bv, acc[j]);
    }
  }

  // Normalize and write out[b, s, hh*64 + d] in fp32.
#pragma unroll
  for (int j = 0; j < 4; ++j)
#pragma unroll
    for (int r = 0; r < 8; ++r) {
      const int s = q0 + r + 8 * h2;
      out[((size_t)b * NS + s) * NDM + hh * ND + 16 * j + n] =
          acc[j][r] / lrow[r];
    }
}

// ---------------------------------------------------------------------------
extern "C" void kernel_launch(void* const* d_in, const int* in_sizes, int n_in,
                              void* d_out, int out_size, void* d_ws,
                              size_t ws_size, hipStream_t stream) {
  const float* Q    = (const float*)d_in[0];
  const float* K    = (const float*)d_in[1];
  const float* V    = (const float*)d_in[2];
  const int*   Vlen = (const int*)d_in[3];
  const float* WQ   = (const float*)d_in[4];
  const float* WK   = (const float*)d_in[5];
  const float* WV   = (const float*)d_in[6];
  float* out = (float*)d_out;

  const size_t elems = (size_t)NB * NH * NS * ND;  // 4,194,304 per tensor
  _Float16* qh = (_Float16*)d_ws;
  _Float16* kh = qh + elems;
  _Float16* vT = kh + elems;

  // Projections: 3 GEMMs, V stored transposed.
  proj_qkv<<<dim3(64, 64, 3), 128, 0, stream>>>(Q, K, V, WQ, WK, WV, qh, kh, vT);
  // Flash attention: 4096 query tiles, 4 waves per block.
  attn_fwd<<<dim3(1024, 1, 1), 128, 0, stream>>>(qh, kh, vT, Vlen, out);
}
